// SphericalLinear_472446403136
// MI455X (gfx1250) — matrix operand order repaired and checked by reading, not compiled, not run on hardware
//
#include <hip/hip_runtime.h>

#define NFILT 4
#define C_IN  256
#define C_OUT 512

typedef __attribute__((ext_vector_type(16))) __bf16       v16bf;
typedef __attribute__((ext_vector_type(8)))  float        v8f;
typedef __attribute__((ext_vector_type(4)))  unsigned int v4u;

// ---------- bf16 split helpers (round-to-nearest-even) ----------
__device__ __forceinline__ unsigned short f32_to_bf16_rne(float x) {
    unsigned int u = __float_as_uint(x);
    u += 0x7FFFu + ((u >> 16) & 1u);
    return (unsigned short)(u >> 16);
}
__device__ __forceinline__ float bf16_hi_f32(unsigned short h) {
    return __uint_as_float(((unsigned int)h) << 16);
}
// split two floats into packed (hi,hi) and (lo,lo) bf16 dwords
__device__ __forceinline__ void split2(float a, float b,
                                       unsigned int& hi, unsigned int& lo) {
    unsigned short ha = f32_to_bf16_rne(a), hb = f32_to_bf16_rne(b);
    unsigned short la = f32_to_bf16_rne(a - bf16_hi_f32(ha));
    unsigned short lb = f32_to_bf16_rne(b - bf16_hi_f32(hb));
    hi = (unsigned int)ha | ((unsigned int)hb << 16);
    lo = (unsigned int)la | ((unsigned int)lb << 16);
}

// ---------- 1. zero bucket counters ----------
__global__ void sl_init_kernel(int* counters) {
    if (threadIdx.x < 12) counters[threadIdx.x] = 0;   // counts[4], offs[4], cursors[4]
}

// ---------- 2. classify: radius bucket per point + histogram ----------
__global__ void sl_classify_kernel(const float* __restrict__ xyz,
                                   int* __restrict__ fid,
                                   int* __restrict__ counts, int P) {
    int p = blockIdx.x * blockDim.x + threadIdx.x;
    if (p >= P) return;
    float x = xyz[3 * p + 0], y = xyz[3 * p + 1], z = xyz[3 * p + 2];
    float r2 = x * x + y * y + z * z;
    int id;                       // first radius with r < R; all-false -> argmax==0
    if      (r2 < 1.0f)     id = 0;
    else if (r2 < 2.25f)    id = 1;
    else if (r2 < 4.0f)     id = 2;
    else if (r2 < 10000.0f) id = 3;
    else                    id = 0;
    fid[p] = id;
    atomicAdd(&counts[id], 1);
}

// ---------- 3. scan: bucket offsets + cursors ----------
__global__ void sl_scan_kernel(int* counters) {
    int* counts = counters; int* offs = counters + 4; int* cur = counters + 8;
    int acc = 0;
    for (int i = 0; i < NFILT; ++i) { offs[i] = acc; cur[i] = acc; acc += counts[i]; }
}

// ---------- 4. scatter: permutation grouping points by bucket ----------
__global__ void sl_scatter_kernel(const int* __restrict__ fid,
                                  int* __restrict__ cursors,
                                  int* __restrict__ perm, int P) {
    int p = blockIdx.x * blockDim.x + threadIdx.x;
    if (p >= P) return;
    int pos = atomicAdd(&cursors[fid[p]], 1);
    perm[pos] = p;
}

// ---------- 5. weights f32 -> bf16 hi/lo ----------
__global__ void sl_convw_kernel(const float* __restrict__ w,
                                unsigned short* __restrict__ w_hi,
                                unsigned short* __restrict__ w_lo, int n) {
    int i = blockIdx.x * blockDim.x + threadIdx.x;
    if (i >= n) return;
    float v = w[i];
    unsigned short h = f32_to_bf16_rne(v);
    w_hi[i] = h;
    w_lo[i] = f32_to_bf16_rne(v - bf16_hi_f32(h));
}

// ---------- 6. bucketed WMMA GEMM ----------
// block = 256 threads (8 waves). grid.x over 16-row tiles across buckets,
// grid.y over column halves (256 cols each). Each wave: two 16x16 N-tiles,
// K = 256 in chunks of 32, 3-term bf16 split accumulation in f32.
#define LDSROW 264   // 256 + 8 ushort pad (16B-aligned rows, stride 132 dwords -> bank-conflict free)

__global__ __launch_bounds__(256) void sl_gemm_kernel(
    const float* __restrict__ feat, const float* __restrict__ bias,
    const unsigned short* __restrict__ w_hi, const unsigned short* __restrict__ w_lo,
    const int* __restrict__ counts, const int* __restrict__ offs,
    const int* __restrict__ perm, float* __restrict__ out) {

    __shared__ unsigned short sAhi[16][LDSROW];
    __shared__ unsigned short sAlo[16][LDSROW];
    __shared__ int sRow[16];
    __shared__ int sF, sTB, sNV;

    if (threadIdx.x == 0) {
        int t = blockIdx.x, f = -1, tb = 0, nv = 0;
        for (int i = 0; i < NFILT; ++i) {
            int c  = counts[i];
            int nt = (c + 15) >> 4;
            if (t < nt) { f = i; tb = offs[i] + t * 16; nv = c - t * 16; break; }
            t -= nt;
        }
        sF = f; sTB = tb; sNV = nv;
    }
    __syncthreads();
    int f = sF;
    if (f < 0) return;                       // uniform exit for surplus blocks

    if (threadIdx.x < 16) {
        int m = threadIdx.x;
        sRow[m] = (m < sNV) ? perm[sTB + m] : -1;
    }
    __syncthreads();

    // ---- stage gathered A tile (16 x 256) as bf16 hi/lo into LDS ----
    {
        int m  = threadIdx.x >> 4;           // row 0..15
        int c0 = (threadIdx.x & 15) << 4;    // 16 consecutive channels
        int p  = sRow[m];
        if (p >= 0) {
            const float4* src = (const float4*)(feat + (size_t)p * C_IN + c0);
            #pragma unroll
            for (int j = 0; j < 4; ++j) {
                float4 v = src[j];
                unsigned int h0, l0, h1, l1;
                split2(v.x, v.y, h0, l0);
                split2(v.z, v.w, h1, l1);
                *(unsigned int*)&sAhi[m][c0 + 4 * j + 0] = h0;
                *(unsigned int*)&sAhi[m][c0 + 4 * j + 2] = h1;
                *(unsigned int*)&sAlo[m][c0 + 4 * j + 0] = l0;
                *(unsigned int*)&sAlo[m][c0 + 4 * j + 2] = l1;
            }
        } else {
            #pragma unroll
            for (int j = 0; j < 8; ++j) {
                *(unsigned int*)&sAhi[m][c0 + 2 * j] = 0u;
                *(unsigned int*)&sAlo[m][c0 + 2 * j] = 0u;
            }
        }
    }
    __syncthreads();

    int lane = threadIdx.x & 31;
    int wave = threadIdx.x >> 5;
    int nbase = blockIdx.y * 256 + wave * 32;     // this wave covers cols [nbase, nbase+32)
    int mrow  = lane & 15;                        // A fragment row
    int kA    = (lane >> 4) << 4;                 // A: 0 or 16  (K sub-group)
    int kB    = (lane >> 4) << 4;                 // B: 0 or 16  (K sub-group)
    int ncol  = lane & 15;                        // B/D column within tile

    size_t wrow0 = ((size_t)f * C_OUT + nbase + ncol) * C_IN;
    size_t wrow1 = wrow0 + (size_t)16 * C_IN;

    union Frag { v4u q[2]; v16bf v; };
    v8f acc0 = {}; v8f acc1 = {};

    #pragma unroll
    for (int kc = 0; kc < C_IN / 32; ++kc) {
        int k0 = kc * 32;
        Frag ah, al, bh0, bl0, bh1, bl1;
        // A fragments from LDS: lane<16 -> K {0-15}; lane>=16 -> K {16-31}
        ah.q[0] = *(const v4u*)&sAhi[mrow][k0 + kA];
        ah.q[1] = *(const v4u*)&sAhi[mrow][k0 + kA + 8];
        al.q[0] = *(const v4u*)&sAlo[mrow][k0 + kA];
        al.q[1] = *(const v4u*)&sAlo[mrow][k0 + kA + 8];
        // B fragments from global bf16 weights: 16 consecutive K per lane
        const v4u* pb;
        pb = (const v4u*)(w_hi + wrow0 + k0 + kB); bh0.q[0] = pb[0]; bh0.q[1] = pb[1];
        pb = (const v4u*)(w_lo + wrow0 + k0 + kB); bl0.q[0] = pb[0]; bl0.q[1] = pb[1];
        pb = (const v4u*)(w_hi + wrow1 + k0 + kB); bh1.q[0] = pb[0]; bh1.q[1] = pb[1];
        pb = (const v4u*)(w_lo + wrow1 + k0 + kB); bl1.q[0] = pb[0]; bl1.q[1] = pb[1];

        // 3-term split-precision accumulation (drop lo*lo)
        acc0 = __builtin_amdgcn_wmma_f32_16x16x32_bf16(false, ah.v, false, bh0.v, (short)0, acc0, false, false);
        acc0 = __builtin_amdgcn_wmma_f32_16x16x32_bf16(false, ah.v, false, bl0.v, (short)0, acc0, false, false);
        acc0 = __builtin_amdgcn_wmma_f32_16x16x32_bf16(false, al.v, false, bh0.v, (short)0, acc0, false, false);
        acc1 = __builtin_amdgcn_wmma_f32_16x16x32_bf16(false, ah.v, false, bh1.v, (short)0, acc1, false, false);
        acc1 = __builtin_amdgcn_wmma_f32_16x16x32_bf16(false, ah.v, false, bl1.v, (short)0, acc1, false, false);
        acc1 = __builtin_amdgcn_wmma_f32_16x16x32_bf16(false, al.v, false, bh1.v, (short)0, acc1, false, false);
    }

    // ---- bias + scatter back through permutation ----
    float bv0 = bias[f * C_OUT + nbase + ncol];
    float bv1 = bias[f * C_OUT + nbase + 16 + ncol];
    int mofs = (lane >> 4) << 3;                 // D: lanes>=16 hold rows M+8
    #pragma unroll
    for (int v = 0; v < 8; ++v) {
        int m = v + mofs;
        int p = sRow[m];
        if (p >= 0) {
            float* o = out + (size_t)p * C_OUT + nbase + ncol;
            o[0]  = acc0[v] + bv0;
            o[16] = acc1[v] + bv1;
        }
    }
}

// ---------- host launcher ----------
extern "C" void kernel_launch(void* const* d_in, const int* in_sizes, int n_in,
                              void* d_out, int out_size, void* d_ws, size_t ws_size,
                              hipStream_t stream) {
    const float* feat   = (const float*)d_in[0];
    const float* xyz    = (const float*)d_in[1];
    const float* weight = (const float*)d_in[2];
    const float* bias   = (const float*)d_in[3];
    float* out = (float*)d_out;

    int P = in_sizes[1] / 3;                     // points = |xyz| / 3

    // workspace layout
    int* counters = (int*)d_ws;                  // counts[4] | offs[4] | cursors[4] | pad
    int* fid      = counters + 16;               // P ints
    int* perm     = fid + P;                     // P ints
    unsigned short* w_hi = (unsigned short*)(perm + P);
    unsigned short* w_lo = w_hi + (size_t)NFILT * C_OUT * C_IN;

    sl_init_kernel<<<1, 32, 0, stream>>>(counters);
    sl_classify_kernel<<<(P + 255) / 256, 256, 0, stream>>>(xyz, fid, counters, P);
    sl_scan_kernel<<<1, 1, 0, stream>>>(counters);
    sl_scatter_kernel<<<(P + 255) / 256, 256, 0, stream>>>(fid, counters + 8, perm, P);

    int nw = NFILT * C_OUT * C_IN;
    sl_convw_kernel<<<(nw + 255) / 256, 256, 0, stream>>>(weight, w_hi, w_lo, nw);

    dim3 grid((P + 15) / 16 + NFILT, C_OUT / 256);   // tile slack for bucket rounding
    sl_gemm_kernel<<<grid, 256, 0, stream>>>(feat, bias, w_hi, w_lo,
                                             counters, counters + 4, perm, out);
}
